// ParallelDroplessMLP_13958643712092
// MI455X (gfx1250) — matrix operand order repaired and checked by reading, not compiled, not run on hardware
//
#include <hip/hip_runtime.h>
#include <hip/hip_bf16.h>

// ---------------------------------------------------------------------------
// MoE dropless MLP (SwiGLU, top-2, 8 experts) for gfx1250 using bf16 WMMA.
// ---------------------------------------------------------------------------

typedef __bf16 bf16;
typedef __attribute__((ext_vector_type(16))) __bf16 bf16x16;
typedef __attribute__((ext_vector_type(8)))  __bf16 bf16x8;
typedef __attribute__((ext_vector_type(4)))  __bf16 bf16x4;
typedef __attribute__((ext_vector_type(8)))  float  f32x8;
typedef __attribute__((ext_vector_type(4)))  float  f32x4;

constexpr int D_MODEL = 1024;
constexpr int HIDDEN  = 4096;
constexpr int NE      = 8;
constexpr int NTOK    = 8192;
constexpr int TROWS   = NTOK * 2;         // 16384 routed rows (top-2)
constexpr int TILE_M  = 64;
constexpr int MAX_TILES = TROWS / TILE_M + NE;   // 264 worst case

// ---- workspace layout ----
// ints: [0..7] counts, [16..24] offsets, [32..39] cursors, [40] ntiles
//       int4 tiles @ byte 256; row_to_slot @ int 4096; slot_to_row @ int 20480
constexpr size_t XG_OFF  = 1ull << 20;                              // bf16 Xg (TROWS x 1024)
constexpr size_t W1T_OFF = XG_OFF  + (size_t)TROWS * D_MODEL * 2;
constexpr size_t W3T_OFF = W1T_OFF + (size_t)NE * HIDDEN * D_MODEL * 2;
constexpr size_t W2T_OFF = W3T_OFF + (size_t)NE * HIDDEN * D_MODEL * 2;
constexpr size_t H_OFF   = W2T_OFF + (size_t)NE * D_MODEL * HIDDEN * 2;
constexpr size_t Y_OFF   = H_OFF   + (size_t)TROWS * HIDDEN * 2;    // f32 Y (TROWS x 1024)

// ---------------------------------------------------------------------------
__global__ void moe_init(int* wsi) {
    int t = threadIdx.x;
    if (t < NE) { wsi[t] = 0; wsi[32 + t] = 0; }
}

__global__ void moe_hist(const int* __restrict__ ei, int* __restrict__ counts) {
    int t = blockIdx.x * blockDim.x + threadIdx.x;
    if (t < TROWS) atomicAdd(&counts[ei[t] & 7], 1);
}

__global__ void moe_scan(int* wsi, float* outTail, int writeTail) {
    if (threadIdx.x != 0) return;
    int* counts  = wsi;
    int* offsets = wsi + 16;
    int4* tiles  = (int4*)((char*)wsi + 256);
    int off = 0, nt = 0;
    for (int e = 0; e < NE; ++e) {
        offsets[e] = off;
        int c = counts[e];
        for (int s = 0; s < c; s += TILE_M) {
            int hi = (s + TILE_M < c) ? (s + TILE_M) : c;
            tiles[nt++] = make_int4(off + s, off + hi, e, 0);
        }
        off += c;
        if (writeTail) outTail[e] = (float)c;
    }
    offsets[NE] = off;
    wsi[40] = nt;
}

__global__ void moe_assign(const int* __restrict__ ei, int* wsi) {
    int t = blockIdx.x * blockDim.x + threadIdx.x;
    if (t >= TROWS) return;
    int e = ei[t] & 7;
    int r = wsi[16 + e] + atomicAdd(&wsi[32 + e], 1);
    wsi[4096 + r]  = t;   // row_to_slot
    wsi[20480 + t] = r;   // slot_to_row
}

// transpose + f32->bf16 convert: src (R x C) f32  ->  dst (C x R) bf16, per z
__global__ __launch_bounds__(256) void moe_tconv(const float* __restrict__ src,
                                                 bf16* __restrict__ dst, int R, int C) {
    __shared__ float tile[32][33];
    size_t zoff = (size_t)blockIdx.z * R * C;
    int c0 = blockIdx.x * 32, r0 = blockIdx.y * 32;
    #pragma unroll
    for (int i = threadIdx.y; i < 32; i += 8)
        tile[i][threadIdx.x] = src[zoff + (size_t)(r0 + i) * C + c0 + threadIdx.x];
    __syncthreads();
    #pragma unroll
    for (int i = threadIdx.y; i < 32; i += 8)
        dst[zoff + (size_t)(c0 + i) * R + r0 + threadIdx.x] = (bf16)tile[threadIdx.x][i];
}

// gather token rows into expert-grouped bf16 activation matrix
__global__ __launch_bounds__(256) void moe_gather(const float* __restrict__ x,
                                                  const int* __restrict__ row_to_slot,
                                                  bf16* __restrict__ Xg) {
    int r = blockIdx.x;
    int token = row_to_slot[r] >> 1;   // slot / TOPK
    int d = threadIdx.x * 4;
    f32x4 v = *(const f32x4*)(x + (size_t)token * D_MODEL + d);
    bf16x4 o;
    o[0] = (bf16)v[0]; o[1] = (bf16)v[1]; o[2] = (bf16)v[2]; o[3] = (bf16)v[3];
    *(bf16x4*)(Xg + (size_t)r * D_MODEL + d) = o;
}

// ---- WMMA fragment loaders (layouts per ISA 7.12.2) ------------------------
// A 16x32 bf16: lane L holds row M=L&15; K = (L>=16?8:0)+j (j<8), +16 for j>=8
__device__ inline bf16x16 load_afrag(const bf16* base, int row, int ld, int k0, int half) {
    const bf16* p = base + (size_t)row * ld + k0 + half * 8;
    bf16x8 v0 = *(const bf16x8*)p;
    bf16x8 v1 = *(const bf16x8*)(p + 16);
    bf16x16 a;
    #pragma unroll
    for (int j = 0; j < 8; ++j) { a[j] = v0[j]; a[8 + j] = v1[j]; }
    return a;
}
// B 32x16 bf16 from N-major (transposed) weights: lane L holds col N=L&15;
// K = (L>=16?8:0)+j (j<8), +16 for j>=8.
__device__ inline bf16x16 load_bfrag(const bf16* wt, int n, int ld, int k0, int half) {
    const bf16* p = wt + (size_t)n * ld + k0 + half * 8;
    bf16x8 v0 = *(const bf16x8*)p;
    bf16x8 v1 = *(const bf16x8*)(p + 16);
    bf16x16 b;
    #pragma unroll
    for (int j = 0; j < 8; ++j) { b[j] = v0[j]; b[8 + j] = v1[j]; }
    return b;
}

__device__ inline float fast_silu_mul(float g, float u) {
    // silu(g)*u with v_rcp_f32 instead of IEEE divide (bf16-output precision)
    return g * __builtin_amdgcn_rcpf(1.0f + __expf(-g)) * u;
}

// ---- GEMM1: H = silu(Xg @ W1e) * (Xg @ W3e), bf16 in/out, f32 accum --------
// block = 256 thr (8 waves, 2x4), tile 64(M) x 128(N), K = 1024
__global__ __launch_bounds__(256) void moe_gemm1(const bf16* __restrict__ Xg,
                                                 const bf16* __restrict__ W1t,
                                                 const bf16* __restrict__ W3t,
                                                 bf16* __restrict__ H,
                                                 const int4* __restrict__ tiles,
                                                 const int* __restrict__ ntiles) {
    if ((int)blockIdx.x >= *ntiles) return;
    int4 tl = tiles[blockIdx.x];
    int rowStart = tl.x, rowEnd = tl.y, e = tl.z;
    int wave = threadIdx.x >> 5, lane = threadIdx.x & 31;
    int half = lane >> 4, l16 = lane & 15;
    int wm = wave >> 2, wn = wave & 3;
    int mBase = rowStart + wm * 32;
    int nBase = blockIdx.y * 128 + wn * 32;

    const bf16* w1e = W1t + (size_t)e * HIDDEN * D_MODEL;
    const bf16* w3e = W3t + (size_t)e * HIDDEN * D_MODEL;

    f32x8 acc1[2][2] = {};
    f32x8 acc3[2][2] = {};

    for (int k0 = 0; k0 < D_MODEL; k0 += 32) {
        if (k0 + 256 < D_MODEL) {   // stream weights ahead (global_prefetch_b8)
            __builtin_prefetch(w1e + (size_t)(nBase + l16) * D_MODEL + k0 + 256, 0, 1);
            __builtin_prefetch(w3e + (size_t)(nBase + l16) * D_MODEL + k0 + 256, 0, 1);
        }
        bf16x16 a[2], b1[2], b3[2];
        #pragma unroll
        for (int mf = 0; mf < 2; ++mf) {
            int r = mBase + mf * 16 + l16;
            r = r < TROWS - 1 ? r : TROWS - 1;          // clamp pad rows in-buffer
            a[mf] = load_afrag(Xg, r, D_MODEL, k0, half);
        }
        #pragma unroll
        for (int nf = 0; nf < 2; ++nf) {
            int n = nBase + nf * 16 + l16;
            b1[nf] = load_bfrag(w1e, n, D_MODEL, k0, half);
            b3[nf] = load_bfrag(w3e, n, D_MODEL, k0, half);
        }
        #pragma unroll
        for (int mf = 0; mf < 2; ++mf)
            #pragma unroll
            for (int nf = 0; nf < 2; ++nf) {
                acc1[mf][nf] = __builtin_amdgcn_wmma_f32_16x16x32_bf16(
                    false, a[mf], false, b1[nf], (short)0, acc1[mf][nf], false, false);
                acc3[mf][nf] = __builtin_amdgcn_wmma_f32_16x16x32_bf16(
                    false, a[mf], false, b3[nf], (short)0, acc3[mf][nf], false, false);
            }
    }

    bool full = (rowEnd - rowStart) == TILE_M;   // uniform over block
    #pragma unroll
    for (int mf = 0; mf < 2; ++mf) {
        int rbase = mBase + mf * 16 + half * 8;  // C layout: VGPR i -> M=i (+8 hi half)
        #pragma unroll
        for (int nf = 0; nf < 2; ++nf) {
            int col = nBase + nf * 16 + l16;
            bf16* hp = H + (size_t)rbase * HIDDEN + col;
            if (full) {
                #pragma unroll
                for (int i = 0; i < 8; ++i)
                    hp[(size_t)i * HIDDEN] =
                        (bf16)fast_silu_mul(acc1[mf][nf][i], acc3[mf][nf][i]);
            } else {
                #pragma unroll
                for (int i = 0; i < 8; ++i)
                    if (rbase + i < rowEnd)
                        hp[(size_t)i * HIDDEN] =
                            (bf16)fast_silu_mul(acc1[mf][nf][i], acc3[mf][nf][i]);
            }
        }
    }
}

// ---- GEMM2: Y = H @ W2e, bf16 in, f32 out; tile 64x128, K = 4096 -----------
__global__ __launch_bounds__(256) void moe_gemm2(const bf16* __restrict__ H,
                                                 const bf16* __restrict__ W2t,
                                                 float* __restrict__ Y,
                                                 const int4* __restrict__ tiles,
                                                 const int* __restrict__ ntiles) {
    if ((int)blockIdx.x >= *ntiles) return;
    int4 tl = tiles[blockIdx.x];
    int rowStart = tl.x, rowEnd = tl.y, e = tl.z;
    int wave = threadIdx.x >> 5, lane = threadIdx.x & 31;
    int half = lane >> 4, l16 = lane & 15;
    int wm = wave >> 2, wn = wave & 3;
    int mBase = rowStart + wm * 32;
    int nBase = blockIdx.y * 128 + wn * 32;

    const bf16* w2e = W2t + (size_t)e * D_MODEL * HIDDEN;

    f32x8 acc[2][2] = {};

    for (int k0 = 0; k0 < HIDDEN; k0 += 32) {
        if (k0 + 256 < HIDDEN) {
            __builtin_prefetch(w2e + (size_t)(nBase + l16) * HIDDEN + k0 + 256, 0, 1);
        }
        bf16x16 a[2], b[2];
        #pragma unroll
        for (int mf = 0; mf < 2; ++mf) {
            int r = mBase + mf * 16 + l16;
            r = r < TROWS - 1 ? r : TROWS - 1;
            a[mf] = load_afrag(H, r, HIDDEN, k0, half);
        }
        #pragma unroll
        for (int nf = 0; nf < 2; ++nf) {
            int n = nBase + nf * 16 + l16;
            b[nf] = load_bfrag(w2e, n, HIDDEN, k0, half);
        }
        #pragma unroll
        for (int mf = 0; mf < 2; ++mf)
            #pragma unroll
            for (int nf = 0; nf < 2; ++nf)
                acc[mf][nf] = __builtin_amdgcn_wmma_f32_16x16x32_bf16(
                    false, a[mf], false, b[nf], (short)0, acc[mf][nf], false, false);
    }

    bool full = (rowEnd - rowStart) == TILE_M;
    #pragma unroll
    for (int mf = 0; mf < 2; ++mf) {
        int rbase = mBase + mf * 16 + half * 8;
        #pragma unroll
        for (int nf = 0; nf < 2; ++nf) {
            int col = nBase + nf * 16 + l16;
            float* yp = Y + (size_t)rbase * D_MODEL + col;
            if (full) {
                #pragma unroll
                for (int i = 0; i < 8; ++i)
                    yp[(size_t)i * D_MODEL] = acc[mf][nf][i];
            } else {
                #pragma unroll
                for (int i = 0; i < 8; ++i)
                    if (rbase + i < rowEnd)
                        yp[(size_t)i * D_MODEL] = acc[mf][nf][i];
            }
        }
    }
}

// ---- combine: out[token] = ew[2t]*Y[row(2t)] + ew[2t+1]*Y[row(2t+1)] -------
__global__ __launch_bounds__(256) void moe_combine(const float* __restrict__ Y,
                                                   const float* __restrict__ ew,
                                                   const int* __restrict__ slot_to_row,
                                                   float* __restrict__ out) {
    int token = blockIdx.x;
    int t0 = token * 2, t1 = t0 + 1;
    int r0 = slot_to_row[t0], r1 = slot_to_row[t1];
    float w0 = ew[t0], w1 = ew[t1];
    int d = threadIdx.x * 4;
    f32x4 y0 = *(const f32x4*)(Y + (size_t)r0 * D_MODEL + d);
    f32x4 y1 = *(const f32x4*)(Y + (size_t)r1 * D_MODEL + d);
    f32x4 v = w0 * y0 + w1 * y1;
    *(f32x4*)(out + (size_t)token * D_MODEL + d) = v;
}

// ---------------------------------------------------------------------------
extern "C" void kernel_launch(void* const* d_in, const int* in_sizes, int n_in,
                              void* d_out, int out_size, void* d_ws, size_t ws_size,
                              hipStream_t stream) {
    const float* x  = (const float*)d_in[0];
    const float* ew = (const float*)d_in[1];
    const int*   ei = (const int*)d_in[2];
    const float* w1 = (const float*)d_in[3];
    const float* w2 = (const float*)d_in[4];
    const float* w3 = (const float*)d_in[5];
    float* out = (float*)d_out;

    char* ws  = (char*)d_ws;
    int*  wsi = (int*)ws;
    bf16* Xg  = (bf16*)(ws + XG_OFF);
    bf16* W1t = (bf16*)(ws + W1T_OFF);
    bf16* W3t = (bf16*)(ws + W3T_OFF);
    bf16* W2t = (bf16*)(ws + W2T_OFF);
    bf16* Hb  = (bf16*)(ws + H_OFF);
    float* Y  = (float*)(ws + Y_OFF);
    const int4* tiles  = (const int4*)(ws + 256);
    const int*  ntiles = wsi + 40;

    int writeTail = (out_size >= NTOK * D_MODEL + NE) ? 1 : 0;
    float* outTail = out + (size_t)NTOK * D_MODEL;

    moe_init  <<<1, 32, 0, stream>>>(wsi);
    moe_hist  <<<TROWS / 256, 256, 0, stream>>>(ei, wsi);
    moe_scan  <<<1, 32, 0, stream>>>(wsi, outTail, writeTail);
    moe_assign<<<TROWS / 256, 256, 0, stream>>>(ei, wsi);

    // weights -> bf16, transposed to N-major (contiguous K per output column)
    moe_tconv<<<dim3(HIDDEN / 32, D_MODEL / 32, NE), dim3(32, 8), 0, stream>>>(w1, W1t, D_MODEL, HIDDEN);
    moe_tconv<<<dim3(HIDDEN / 32, D_MODEL / 32, NE), dim3(32, 8), 0, stream>>>(w3, W3t, D_MODEL, HIDDEN);
    moe_tconv<<<dim3(D_MODEL / 32, HIDDEN / 32, NE), dim3(32, 8), 0, stream>>>(w2, W2t, HIDDEN, D_MODEL);

    moe_gather<<<TROWS, 256, 0, stream>>>(x, wsi + 4096, Xg);

    moe_gemm1<<<dim3(MAX_TILES, HIDDEN / 128), 256, 0, stream>>>(Xg, W1t, W3t, Hb, tiles, ntiles);
    moe_gemm2<<<dim3(MAX_TILES, D_MODEL / 128), 256, 0, stream>>>(Hb, W2t, Y, tiles, ntiles);

    moe_combine<<<NTOK, 256, 0, stream>>>(Y, ew, wsi + 20480, out);
}
